// GraphExpert_81398220194597
// MI455X (gfx1250) — matrix operand order repaired and checked by reading, not compiled, not run on hardware
//
#include <hip/hip_runtime.h>
#include <hip/hip_bf16.h>

#define NODES 16
#define HSZ   512
#define BSZ   2048
#define BM    64      // batch rows per workgroup
#define BN    128     // output-column pass width
#define KT    32      // WMMA K per step
#define NK    (HSZ / KT)
#define NTBUF 3                       // TDM tile buffers (triple buffered)
#define TILEE (BN * KT)               // elements per B tile
#define TILEB (TILEE * 2)             // bytes per B tile

typedef __attribute__((ext_vector_type(16))) __bf16 v16bf;
typedef __attribute__((ext_vector_type(8)))  __bf16 v8bf;
typedef __attribute__((ext_vector_type(8)))  float  v8f;
typedef __attribute__((ext_vector_type(4)))  unsigned int u32x4;
typedef __attribute__((ext_vector_type(8)))  int i32x8;
typedef __attribute__((ext_vector_type(4)))  int i32x4;

union BFrag { v16bf v; v8bf h[2]; };

__device__ __forceinline__ v8f wmma_bf16(v16bf a, v16bf b, v8f c) {
  return __builtin_amdgcn_wmma_f32_16x16x32_bf16(
      /*neg_a=*/false, a, /*neg_b=*/false, b,
      /*c_mod=*/(short)0, c, /*reuse_a=*/false, /*reuse_b=*/false);
}

// A fragment: 16x32 bf16 from a row-major [rows][HSZ] bf16 LDS block.
__device__ __forceinline__ v16bf load_a_frag(const __bf16* rowBase, int lane) {
  const int lo = lane & 15, hi = lane >> 4;
  const __bf16* p = rowBase + (size_t)lo * HSZ + hi * 8;
  BFrag f;
  f.h[0] = *(const v8bf*)(p);        // K = hi*8 .. +7
  f.h[1] = *(const v8bf*)(p + 16);   // K = 16 + hi*8 .. +7
  return f.v;
}

// B fragment: 32x16 (KxN) bf16 from a transposed [BN cols][KT] bf16 LDS tile.
__device__ __forceinline__ v16bf load_b_frag(const __bf16* Bt, int colBase, int lane) {
  const int lo = lane & 15, hi = lane >> 4;
  const __bf16* p = Bt + (size_t)(colBase + lo) * KT + hi * 8;
  BFrag f;
  f.h[0] = *(const v8bf*)(p);
  f.h[1] = *(const v8bf*)(p + 16);
  return f.v;
}

// Tensor Data Mover: DMA a [BN cols][KT k] bf16 tile (source is column-major
// Wt[node][col][k]) into LDS at ldsOff. 2-D descriptor: tile_dim0 = KT
// contiguous k-elements, tile_dim1 = BN rows, tensor_dim0_stride = HSZ.
__device__ __forceinline__ void tdm_load_tile(const __bf16* gsrc, unsigned ldsOff) {
  const unsigned long long ga = (unsigned long long)(size_t)gsrc;
  u32x4 g0 = { 1u,                                    // count=1 (valid descriptor)
               ldsOff,                                // lds_addr (bytes)
               (unsigned)ga,                          // global_addr[31:0]
               ((unsigned)(ga >> 32) & 0x01FFFFFFu) | 0x80000000u }; // addr[56:32] | type=2
  i32x8 g1 = { 0x00010000,                            // data_size=1 (2B); mask=0
               (int)(HSZ << 16),                      // tensor_dim0 = 512 (bits 79:48 lo)
               0x20000000,                            // tensor_dim1 = 8192 (bits 111:80 lo16)
               (int)(KT << 16),                       // tile_dim0 = 32 (bits 127:112)
               BN,                                    // tile_dim1 = 128, tile_dim2 = 0
               HSZ,                                   // tensor_dim0_stride = 512 (lo32)
               0, 0 };
  i32x4 gz4 = { 0, 0, 0, 0 };
#if defined(__clang_major__) && (__clang_major__ >= 23)
  i32x8 gz8 = { 0, 0, 0, 0, 0, 0, 0, 0 };
  __builtin_amdgcn_tensor_load_to_lds(g0, g1, gz4, gz4, gz8, 0);
#else
  __builtin_amdgcn_tensor_load_to_lds(g0, g1, gz4, gz4, 0);
#endif
}

// One GEMM pass: acc(2x2 frags, 32x32 per wave) = Asrc[BMxHSZ] * WtN[cb:cb+BN][:]^T
// WtN is this node's column-major bf16 weight block [HSZ cols][HSZ k].
// B tiles stream through 3 LDS buffers via TDM; 2 DMAs in flight; one barrier
// per K-step (the head barrier proves both "tile ks ready" and "buffer for
// tile ks+2 free", since that buffer was last read in iteration ks-1).
__device__ __forceinline__ void gemm_pass(const __bf16* Asrc, const __bf16* __restrict__ WtN,
                                          __bf16* BtBase, int cb,
                                          int wave, int wm, int wn, int lane,
                                          v8f acc[2][2]) {
  const v8f z = {0.f, 0.f, 0.f, 0.f, 0.f, 0.f, 0.f, 0.f};
  acc[0][0] = z; acc[0][1] = z; acc[1][0] = z; acc[1][1] = z;

  const unsigned ldsBase = (unsigned)(size_t)BtBase;
  const __bf16* src = WtN + (size_t)cb * HSZ;

  __syncthreads();                     // previous users of all Bt buffers are done
  if (wave == 0) {                     // preload tiles 0 and 1
    tdm_load_tile(src,      ldsBase);
    tdm_load_tile(src + KT, ldsBase + TILEB);
  }

  for (int ks = 0; ks < NK; ++ks) {
    if (wave == 0) {
      if (ks < NK - 1) __builtin_amdgcn_s_wait_tensorcnt(1);  // tile ks done, ks+1 in flight
      else             __builtin_amdgcn_s_wait_tensorcnt(0);
    }
    __syncthreads();                   // tile ks visible; all waves past iter ks-1 reads

    if (wave == 0 && ks + 2 < NK)      // buffer (ks+2)%3 last read in iter ks-1: free now
      tdm_load_tile(src + (size_t)(ks + 2) * KT, ldsBase + ((ks + 2) % NTBUF) * TILEB);

    const __bf16* cur = BtBase + (ks % NTBUF) * TILEE;
    const v16bf a0  = load_a_frag(Asrc + (size_t)(wm * 32 +  0) * HSZ + ks * KT, lane);
    const v16bf a1  = load_a_frag(Asrc + (size_t)(wm * 32 + 16) * HSZ + ks * KT, lane);
    const v16bf bf0 = load_b_frag(cur, wn * 32 +  0, lane);
    const v16bf bf1 = load_b_frag(cur, wn * 32 + 16, lane);

    acc[0][0] = wmma_bf16(a0, bf0, acc[0][0]);
    acc[0][1] = wmma_bf16(a0, bf1, acc[0][1]);
    acc[1][0] = wmma_bf16(a1, bf0, acc[1][0]);
    acc[1][1] = wmma_bf16(a1, bf1, acc[1][1]);
    // no trailing barrier: next iteration's head barrier provides it
  }
}

// Fused per-node expert MLP + probability combine.
// grid: (BSZ/BM, NODES), block: 256 (8 waves, 2(M) x 4(N))
__global__ void __launch_bounds__(256)
expert_kernel(const float* __restrict__ graphs,
              const __bf16* __restrict__ Wt1, const float* __restrict__ b1,
              const __bf16* __restrict__ Wt2, const float* __restrict__ b2,
              const float* __restrict__ probs, float* __restrict__ combined) {
  extern __shared__ char smem[];
  __bf16* Ablk = (__bf16*)(smem);                          // BM x HSZ bf16 (64 KB)
  __bf16* Hblk = (__bf16*)(smem + BM * HSZ * 2);           // BM x HSZ bf16 (64 KB)
  __bf16* BtB  = (__bf16*)(smem + 2 * BM * HSZ * 2);       // 3 x BN x KT bf16 (24 KB)
  float*  Pblk = (float*)(smem + 2 * BM * HSZ * 2 + NTBUF * TILEB);          // 4 KB

  const int tid  = threadIdx.x;
  const int lane = tid & 31;
  const int wave = tid >> 5;
  const int wm = wave >> 2, wn = wave & 3;
  const int lo = lane & 15, hi = lane >> 4;
  const int node = blockIdx.y;
  const int b0   = blockIdx.x * BM;

  // Stage graphs[b0:b0+BM, node, :] -> Ablk as bf16 (one-time per workgroup)
#pragma unroll
  for (int i = 0; i < (BM * HSZ / 4) / 256; ++i) {         // 32 iterations
    const int c   = tid + i * 256;
    const int row = c >> 7;
    const int c4  = c & 127;
    const float4 v = *(const float4*)(graphs + (((size_t)(b0 + row)) * NODES + node) * HSZ + c4 * 4);
    __bf16* d = Ablk + (size_t)row * HSZ + c4 * 4;
    d[0] = (__bf16)v.x; d[1] = (__bf16)v.y; d[2] = (__bf16)v.z; d[3] = (__bf16)v.w;
  }
#pragma unroll
  for (int i = 0; i < (BM * NODES) / 256; ++i) {
    const int idx = tid + i * 256;
    Pblk[idx] = probs[(size_t)b0 * NODES + idx];
  }
  // note: gemm_pass begins with __syncthreads(), covering the staging above.

  const __bf16* W1n = Wt1 + (size_t)node * HSZ * HSZ;
  const __bf16* W2n = Wt2 + (size_t)node * HSZ * HSZ;
  v8f acc[2][2];

  // ---- Phase 1: Hblk = relu(Ablk * W1[node] + b1[node]) (stays in LDS) ----
  for (int p = 0; p < HSZ / BN; ++p) {
    const int cb = p * BN;
    gemm_pass(Ablk, W1n, BtB, cb, wave, wm, wn, lane, acc);
#pragma unroll
    for (int mt = 0; mt < 2; ++mt)
#pragma unroll
      for (int nt = 0; nt < 2; ++nt) {
        const int col   = cb + wn * 32 + nt * 16 + lo;
        const float bia = b1[(size_t)node * HSZ + col];
#pragma unroll
        for (int r = 0; r < 8; ++r) {
          const int m = wm * 32 + mt * 16 + r + 8 * hi;   // C layout: (m,n)=(r+8*hi, lo)
          float v = acc[mt][nt][r] + bia;
          v = fmaxf(v, 0.f);
          Hblk[(size_t)m * HSZ + col] = (__bf16)v;
        }
      }
  }
  // Hblk writes are published by the head barrier of the next gemm_pass.

  // ---- Phase 2: outs = Hblk * W2[node] + b2[node]; weighted combine ----
  // combined[b, 32*node + o/16] = sum_k probs[b,k]*outs[b, o-grp + k] over 16 cols.
  for (int p = 0; p < HSZ / BN; ++p) {
    const int cb = p * BN;
    gemm_pass(Hblk, W2n, BtB, cb, wave, wm, wn, lane, acc);
#pragma unroll
    for (int mt = 0; mt < 2; ++mt)
#pragma unroll
      for (int nt = 0; nt < 2; ++nt) {
        const int cB    = cb + wn * 32 + nt * 16;          // 16-aligned column group
        const float bia = b2[(size_t)node * HSZ + cB + lo];
        const int j     = 32 * node + (cB >> 4);           // combined column
        float s[8];
#pragma unroll
        for (int r = 0; r < 8; ++r) {
          const int m   = wm * 32 + mt * 16 + r + 8 * hi;  // local batch row
          const float w = Pblk[m * NODES + lo];            // probs[b, k], k = col%16
          s[r] = (acc[mt][nt][r] + bia) * w;
        }
#pragma unroll
        for (int off = 8; off >= 1; off >>= 1)
#pragma unroll
          for (int r = 0; r < 8; ++r)
            s[r] += __shfl_xor(s[r], off, 32);
        if (lo == 0) {
#pragma unroll
          for (int r = 0; r < 8; ++r) {
            const int m = b0 + wm * 32 + mt * 16 + r + 8 * hi;
            combined[(size_t)m * HSZ + j] = s[r];
          }
        }
      }
  }
}

// Convert + transpose W[n][k][c] f32 -> Wt[n][c][k] bf16, 64x64 LDS tiles.
// grid: (HSZ/64, HSZ/64, 2*NODES); z encodes {matrix, node}.
__global__ void __launch_bounds__(256)
prep_kernel(const float* __restrict__ W1, const float* __restrict__ W2,
            __bf16* __restrict__ Wt1, __bf16* __restrict__ Wt2) {
  __shared__ __bf16 tile[64][80];    // 80 => 160B row stride (16B aligned)
  const int mat  = blockIdx.z >> 4;
  const int node = blockIdx.z & 15;
  const float* W = mat ? W2 : W1;
  __bf16* Wt     = mat ? Wt2 : Wt1;
  const int k0 = blockIdx.y * 64;
  const int c0 = blockIdx.x * 64;
  const int tid = threadIdx.x;
#pragma unroll
  for (int i = 0; i < 4; ++i) {                  // 64 k-rows x 16 float4
    const int ch = tid + i * 256;
    const int kr = ch >> 4;
    const int c4 = ch & 15;
    const float4 v = *(const float4*)(W + (((size_t)node * HSZ) + (k0 + kr)) * HSZ + c0 + c4 * 4);
    tile[c4 * 4 + 0][kr] = (__bf16)v.x;
    tile[c4 * 4 + 1][kr] = (__bf16)v.y;
    tile[c4 * 4 + 2][kr] = (__bf16)v.z;
    tile[c4 * 4 + 3][kr] = (__bf16)v.w;
  }
  __syncthreads();
#pragma unroll
  for (int i = 0; i < 2; ++i) {                  // 64 c-rows x 8 chunks of 8 bf16
    const int ch = tid + i * 256;
    const int cr = ch >> 3;
    const int k8 = (ch & 7) * 8;
    const v8bf val = *(const v8bf*)(&tile[cr][k8]);
    *(v8bf*)(Wt + (((size_t)node * HSZ) + (c0 + cr)) * HSZ + k0 + k8) = val;
  }
}

// Gating: logits = graphs_flat @ Wg + bg; probs = softmax(logits). One block per row.
__global__ void __launch_bounds__(256)
gate_kernel(const float* __restrict__ graphs, const float* __restrict__ Wg,
            const float* __restrict__ bg, float* __restrict__ probs) {
  const int b = blockIdx.x;
  const int tid = threadIdx.x;
  const int lane = tid & 31, wave = tid >> 5;
  const float* g = graphs + (size_t)b * (NODES * HSZ);

  float acc[NODES];
#pragma unroll
  for (int e = 0; e < NODES; ++e) acc[e] = 0.f;
  for (int f = tid; f < NODES * HSZ; f += 256) {
    const float gv = g[f];
    const float* w = Wg + (size_t)f * NODES;
#pragma unroll
    for (int e = 0; e < NODES; ++e) acc[e] = fmaf(gv, w[e], acc[e]);
  }
#pragma unroll
  for (int e = 0; e < NODES; ++e)
#pragma unroll
    for (int off = 16; off >= 1; off >>= 1)
      acc[e] += __shfl_xor(acc[e], off, 32);

  __shared__ float red[8][NODES];
  if (lane == 0) {
#pragma unroll
    for (int e = 0; e < NODES; ++e) red[wave][e] = acc[e];
  }
  __syncthreads();

  if (wave == 0) {
    const int e = lane & 15;
    float s = bg[e];
#pragma unroll
    for (int w = 0; w < 8; ++w) s += red[w][e];
    float mx = s;
#pragma unroll
    for (int off = 8; off >= 1; off >>= 1) mx = fmaxf(mx, __shfl_xor(mx, off, 32));
    const float ex = __expf(s - mx);
    float sum = ex;
#pragma unroll
    for (int off = 8; off >= 1; off >>= 1) sum += __shfl_xor(sum, off, 32);
    if (lane < 16) probs[(size_t)b * NODES + e] = ex / sum;
  }
}

extern "C" void kernel_launch(void* const* d_in, const int* in_sizes, int n_in,
                              void* d_out, int out_size, void* d_ws, size_t ws_size,
                              hipStream_t stream) {
  (void)in_sizes; (void)n_in; (void)out_size; (void)ws_size;
  const float* graphs = (const float*)d_in[0];
  const float* W1     = (const float*)d_in[1];
  const float* b1     = (const float*)d_in[2];
  const float* W2     = (const float*)d_in[3];
  const float* b2     = (const float*)d_in[4];
  const float* Wg     = (const float*)d_in[5];
  const float* bg     = (const float*)d_in[6];

  float* out      = (float*)d_out;
  float* combined = out;                            // (2048, 512)
  float* probs    = out + (size_t)BSZ * HSZ;        // (2048, 1, 16) flat

  // Workspace: column-major bf16 weights for TDM tiles (8 MB each).
  __bf16* Wt1 = (__bf16*)d_ws;
  __bf16* Wt2 = Wt1 + (size_t)NODES * HSZ * HSZ;

  gate_kernel<<<dim3(BSZ), dim3(256), 0, stream>>>(graphs, Wg, bg, probs);
  prep_kernel<<<dim3(HSZ / 64, HSZ / 64, 2 * NODES), dim3(256), 0, stream>>>(W1, W2, Wt1, Wt2);

  const size_t smem = (size_t)2 * BM * HSZ * 2      // Ablk + Hblk (bf16)
                    + (size_t)NTBUF * TILEB         // 3 TDM tile buffers
                    + (size_t)BM * NODES * 4;       // Pblk (f32)
  (void)hipFuncSetAttribute((const void*)expert_kernel,
                            hipFuncAttributeMaxDynamicSharedMemorySize, (int)smem);
  expert_kernel<<<dim3(BSZ / BM, NODES), dim3(256), smem, stream>>>(
      graphs, Wt1, b1, Wt2, b2, probs, combined);
}
